// StripTransformerBlock_42649025249312
// MI455X (gfx1250) — matrix operand order repaired and checked by reading, not compiled, not run on hardware
//
#include <hip/hip_runtime.h>
#include <math.h>

typedef __bf16 bf16_t;
typedef __attribute__((ext_vector_type(16))) bf16_t v16bf;
typedef __attribute__((ext_vector_type(8)))  bf16_t v8bf;
typedef __attribute__((ext_vector_type(8)))  float  v8f;

// ---------------- problem constants ----------------
static constexpr int kH    = 128;
static constexpr int kW    = 128;
static constexpr int kC    = 192;
static constexpr int kSW   = 8;
static constexpr int kNH   = 6;
static constexpr int kHD   = 32;
static constexpr int kN    = 1024;   // tokens per strip
static constexpr int kNS   = 16;     // strips
static constexpr int kTOK  = 16384;  // total tokens
static constexpr float kScaleQ = 0.17677669529663687f; // 32^-0.5

// ---------------- WMMA operand loaders (bf16 16x16x32) ----------------
// A (16x32, row-major source, row stride = lda elements):
//   lane L -> row (L&15), K base (L>>4)*8; vector elems 0..7 = K kb..kb+7,
//   elems 8..15 = K kb+16..kb+23  (per ISA 16-bit A layout).
__device__ __forceinline__ v16bf load_a16x32(const bf16_t* base, int lda, int lane) {
    int m  = lane & 15;
    int kb = (lane >> 4) << 3;           // 0 or 8
    const bf16_t* p = base + m * lda + kb;
    v8bf lo = *(const v8bf*)(p);         // K = kb .. kb+7
    v8bf hi = *(const v8bf*)(p + 16);    // K = kb+16 .. kb+23
    return __builtin_shufflevector(lo, hi, 0,1,2,3,4,5,6,7,8,9,10,11,12,13,14,15);
}

// B (32x16) loaded from B^T stored row-major (rows = output columns, stride ldbt):
//   lane L -> column (L&15), K base (L>>4)*8; elems 0..7 = K kb..kb+7, elems 8..15 = K kb+16..kb+23.
__device__ __forceinline__ v16bf load_b32x16(const bf16_t* bt, int ldbt, int lane) {
    int n  = lane & 15;
    int kb = (lane >> 4) << 3;           // 0 or 8
    const bf16_t* p = bt + n * ldbt + kb;
    v8bf lo = *(const v8bf*)(p);
    v8bf hi = *(const v8bf*)(p + 16);
    return __builtin_shufflevector(lo, hi, 0,1,2,3,4,5,6,7,8,9,10,11,12,13,14,15);
}

__device__ __forceinline__ v8f wmma_bf16(v16bf a, v16bf b, v8f c) {
    return __builtin_amdgcn_wmma_f32_16x16x32_bf16(false, a, false, b, (short)0, c, false, false);
}

// 2x2 register-blocked wave tile: 32x32 output, 4 accumulators, each A/B
// fragment reused twice -> 2 b128 loads per WMMA instead of 4.
struct Acc2x2 { v8f m[2][2]; };

template <int KSTEPS>
__device__ __forceinline__ Acc2x2 gemm_2x2(const bf16_t* A, int lda,
                                           const bf16_t* BT, int ldbt, int lane) {
    const v8f z = {0.f,0.f,0.f,0.f,0.f,0.f,0.f,0.f};
    Acc2x2 t;
    t.m[0][0] = z; t.m[0][1] = z; t.m[1][0] = z; t.m[1][1] = z;
#pragma unroll
    for (int ks = 0; ks < KSTEPS; ++ks) {
        v16bf a0 = load_a16x32(A + ks * 32, lda, lane);
        v16bf a1 = load_a16x32(A + 16 * lda + ks * 32, lda, lane);
        v16bf b0 = load_b32x16(BT + ks * 32, ldbt, lane);
        v16bf b1 = load_b32x16(BT + 16 * ldbt + ks * 32, ldbt, lane);
        t.m[0][0] = wmma_bf16(a0, b0, t.m[0][0]);
        t.m[0][1] = wmma_bf16(a0, b1, t.m[0][1]);
        t.m[1][0] = wmma_bf16(a1, b0, t.m[1][0]);
        t.m[1][1] = wmma_bf16(a1, b1, t.m[1][1]);
    }
    return t;
}

// ---------------- weight fp32 -> bf16 convert ----------------
__global__ void cvt_weights(const float* qkv_w, const float* proj_w,
                            const float* fc1_w, const float* fc2_w,
                            bf16_t* wq, bf16_t* wp, bf16_t* w1, bf16_t* w2) {
    int i = blockIdx.x * 256 + threadIdx.x;
    if (i < 576 * 192) wq[i] = (bf16_t)qkv_w[i];
    if (i < 192 * 192) wp[i] = (bf16_t)proj_w[i];
    if (i < 768 * 192) w1[i] = (bf16_t)fc1_w[i];
    if (i < 192 * 768) w2[i] = (bf16_t)fc2_w[i];
}

// ---------------- LayerNorm (wave per token), optional strip permute ----------------
template <int PERMUTE>
__global__ void ln_kernel(const float* __restrict__ x, const float* __restrict__ g,
                          const float* __restrict__ b, bf16_t* __restrict__ out) {
    int wave  = threadIdx.x >> 5;
    int lane  = threadIdx.x & 31;
    int token = blockIdx.x * (blockDim.x >> 5) + wave;
    const float* row = x + token * kC;
    float v[6];
    float s = 0.f;
#pragma unroll
    for (int i = 0; i < 6; ++i) { v[i] = row[lane + i * 32]; s += v[i]; }
#pragma unroll
    for (int m = 1; m < 32; m <<= 1) s += __shfl_xor(s, m, 32);
    float mu = s * (1.f / 192.f);
    float var = 0.f;
#pragma unroll
    for (int i = 0; i < 6; ++i) { float d = v[i] - mu; var += d * d; }
#pragma unroll
    for (int m = 1; m < 32; m <<= 1) var += __shfl_xor(var, m, 32);
    var *= (1.f / 192.f);
    float rs = rsqrtf(var + 1e-5f);

    bf16_t* orow;
    if (PERMUTE) {
        int hh = token >> 7, w = token & 127;
        int st = w >> 3, n = hh * kSW + (w & 7);
        orow = out + (st * kN + n) * kC;
    } else {
        orow = out + token * kC;
    }
#pragma unroll
    for (int i = 0; i < 6; ++i) {
        int c = lane + i * 32;
        orow[c] = (bf16_t)((v[i] - mu) * rs * g[c] + b[c]);
    }
}

// ---------------- QKV GEMM: (16384x192) x (192x576), 32x32 wave tiles ----------------
__global__ void qkv_gemm(const bf16_t* __restrict__ xs, const bf16_t* __restrict__ wq,
                         const float* __restrict__ qkv_b,
                         bf16_t* __restrict__ qb, bf16_t* __restrict__ kb,
                         bf16_t* __restrict__ vtb) {
    int lane = threadIdx.x & 31;
    int wave = threadIdx.x >> 5;
    int tid  = blockIdx.x * 8 + wave;
    const int TN = 576 / 32;              // 18
    int tm = tid / TN, tn = tid % TN;

    Acc2x2 acc = gemm_2x2<6>(xs + tm * 32 * kC, kC, wq + tn * 32 * kC, kC, lane);

    int ncol = lane & 15, hi8 = (lane >> 4) * 8;
#pragma unroll
    for (int ni = 0; ni < 2; ++ni) {
        int col = tn * 32 + ni * 16 + ncol;
        float bias = qkv_b[col];
        int which = col / kC, rem = col % kC;
        int head = rem >> 5, hd = rem & 31;
#pragma unroll
        for (int mi = 0; mi < 2; ++mi) {
#pragma unroll
            for (int r = 0; r < 8; ++r) {
                int row = tm * 32 + mi * 16 + r + hi8;
                int s = row >> 10, n = row & 1023;
                float val = acc.m[mi][ni][r] + bias;
                if (which == 0)
                    qb[((s * kNH + head) * kN + n) * kHD + hd] = (bf16_t)(val * kScaleQ);
                else if (which == 1)
                    kb[((s * kNH + head) * kN + n) * kHD + hd] = (bf16_t)val;
                else
                    vtb[((s * kNH + head) * kHD + hd) * kN + n] = (bf16_t)val;  // V^T
            }
        }
    }
}

// ---------------- fused flash attention per (strip, head) ----------------
// 4 waves / block, each wave owns 16 query rows; streams 32-key blocks.
__global__ void attn_kernel(const bf16_t* __restrict__ qb, const bf16_t* __restrict__ kb,
                            const bf16_t* __restrict__ vtb, const float* __restrict__ rpb,
                            bf16_t* __restrict__ ob) {
    __shared__ __align__(16) bf16_t lds_p[4][16 * 32];
    int lane = threadIdx.x & 31;
    int wave = threadIdx.x >> 5;
    int gid  = blockIdx.x;
    int s    = gid / (kNH * 16);
    int rem  = gid % (kNH * 16);
    int h    = rem / 16;
    int qt   = rem % 16;
    int q0   = (qt * 4 + wave) * 16;

    const bf16_t* qbase = qb  + ((s * kNH + h) * kN + q0) * kHD;
    const bf16_t* kbase = kb  + ((s * kNH + h) * kN) * kHD;
    const bf16_t* vbase = vtb + ((s * kNH + h) * kHD) * kN;
    bf16_t* lds = lds_p[wave];

    v16bf qa = load_a16x32(qbase, kHD, lane);

    const v8f vzero = {0.f,0.f,0.f,0.f,0.f,0.f,0.f,0.f};
    v8f acc0 = vzero, acc1 = vzero;
    float mrow[8], lrow[8];
#pragma unroll
    for (int r = 0; r < 8; ++r) { mrow[r] = -INFINITY; lrow[r] = 0.f; }

    int ncol = lane & 15, hi8 = (lane >> 4) * 8;

    for (int blk = 0; blk < kN / 32; ++blk) {
        int k0 = blk * 32;
        // scores: one WMMA per 16x16 tile (K = HD = 32)
        v16bf b0 = load_b32x16(kbase + k0 * kHD, kHD, lane);
        v16bf b1 = load_b32x16(kbase + (k0 + 16) * kHD, kHD, lane);
        v8f s0 = wmma_bf16(qa, b0, vzero);
        v8f s1 = wmma_bf16(qa, b1, vzero);

        // relative position bias
        int key0 = k0 + ncol, key1 = key0 + 16;
        int chk0 = key0 >> 3, cwk0 = key0 & 7;
        int chk1 = key1 >> 3, cwk1 = key1 & 7;
#pragma unroll
        for (int r = 0; r < 8; ++r) {
            int qrow = q0 + r + hi8;
            int chq = qrow >> 3, cwq = qrow & 7;
            int i0 = (chq - chk0 + kH - 1) * (2 * kSW - 1) + (cwq - cwk0 + kSW - 1);
            int i1 = (chq - chk1 + kH - 1) * (2 * kSW - 1) + (cwq - cwk1 + kSW - 1);
            s0[r] += rpb[i0 * kNH + h];
            s1[r] += rpb[i1 * kNH + h];
        }

        // online softmax: rows live across one 16-lane half -> xor-shuffle reduce
#pragma unroll
        for (int r = 0; r < 8; ++r) {
            float mx = fmaxf(s0[r], s1[r]);
            mx = fmaxf(mx, __shfl_xor(mx, 1, 32));
            mx = fmaxf(mx, __shfl_xor(mx, 2, 32));
            mx = fmaxf(mx, __shfl_xor(mx, 4, 32));
            mx = fmaxf(mx, __shfl_xor(mx, 8, 32));
            float mnew = fmaxf(mrow[r], mx);
            float corr = __expf(mrow[r] - mnew);
            float p0 = __expf(s0[r] - mnew);
            float p1 = __expf(s1[r] - mnew);
            float rs = p0 + p1;
            rs += __shfl_xor(rs, 1, 32);
            rs += __shfl_xor(rs, 2, 32);
            rs += __shfl_xor(rs, 4, 32);
            rs += __shfl_xor(rs, 8, 32);
            lrow[r] = lrow[r] * corr + rs;
            mrow[r] = mnew;
            acc0[r] *= corr;
            acc1[r] *= corr;
            int rr = r + hi8;
            lds[rr * 32 + ncol]      = (bf16_t)p0;  // C-layout -> LDS (row-major 16x32)
            lds[rr * 32 + 16 + ncol] = (bf16_t)p1;
        }

        // O += P(16x32) x V(32x16) per head-dim half; DS ops in-order per wave
        v16bf pa = load_a16x32(lds, 32, lane);
        v16bf v0 = load_b32x16(vbase + k0, kN, lane);              // hd 0..15
        v16bf v1 = load_b32x16(vbase + 16 * kN + k0, kN, lane);    // hd 16..31
        acc0 = wmma_bf16(pa, v0, acc0);
        acc1 = wmma_bf16(pa, v1, acc1);
    }

#pragma unroll
    for (int r = 0; r < 8; ++r) {
        float inv = 1.f / lrow[r];
        int row = q0 + r + hi8;
        bf16_t* op = ob + (s * kN + row) * kC + h * kHD;
        op[ncol]      = (bf16_t)(acc0[r] * inv);
        op[16 + ncol] = (bf16_t)(acc1[r] * inv);
    }
}

// ---------------- proj GEMM + bias + residual + strip reverse ----------------
__global__ void proj_gemm(const bf16_t* __restrict__ ob, const bf16_t* __restrict__ wp,
                          const float* __restrict__ proj_b, const float* __restrict__ x0,
                          float* __restrict__ x1) {
    int lane = threadIdx.x & 31;
    int wave = threadIdx.x >> 5;
    int tid  = blockIdx.x * 8 + wave;
    const int TN = kC / 32;                 // 6
    int tm = tid / TN, tn = tid % TN;

    Acc2x2 acc = gemm_2x2<6>(ob + tm * 32 * kC, kC, wp + tn * 32 * kC, kC, lane);

    int ncol = lane & 15, hi8 = (lane >> 4) * 8;
#pragma unroll
    for (int ni = 0; ni < 2; ++ni) {
        int col = tn * 32 + ni * 16 + ncol;
        float bias = proj_b[col];
#pragma unroll
        for (int mi = 0; mi < 2; ++mi) {
#pragma unroll
            for (int r = 0; r < 8; ++r) {
                int row = tm * 32 + mi * 16 + r + hi8;
                int s = row >> 10, n = row & 1023;
                int hh = n >> 3, wi = n & 7;
                int token = hh * kW + s * kSW + wi;     // strip reverse
                x1[token * kC + col] = x0[token * kC + col] + acc.m[mi][ni][r] + bias;
            }
        }
    }
}

// ---------------- FC1 GEMM + exact GELU ----------------
__global__ void fc1_gemm(const bf16_t* __restrict__ xn2, const bf16_t* __restrict__ w1,
                         const float* __restrict__ fc1_b, bf16_t* __restrict__ h1) {
    int lane = threadIdx.x & 31;
    int wave = threadIdx.x >> 5;
    int tid  = blockIdx.x * 8 + wave;
    const int TN = 768 / 32;                // 24
    int tm = tid / TN, tn = tid % TN;

    Acc2x2 acc = gemm_2x2<6>(xn2 + tm * 32 * kC, kC, w1 + tn * 32 * kC, kC, lane);

    int ncol = lane & 15, hi8 = (lane >> 4) * 8;
#pragma unroll
    for (int ni = 0; ni < 2; ++ni) {
        int col = tn * 32 + ni * 16 + ncol;
        float bias = fc1_b[col];
#pragma unroll
        for (int mi = 0; mi < 2; ++mi) {
#pragma unroll
            for (int r = 0; r < 8; ++r) {
                int row = tm * 32 + mi * 16 + r + hi8;
                float t = acc.m[mi][ni][r] + bias;
                float ge = 0.5f * t * (1.f + erff(t * 0.70710678118654752f));
                h1[row * 768 + col] = (bf16_t)ge;
            }
        }
    }
}

// ---------------- FC2 GEMM + bias + residual -> output ----------------
__global__ void fc2_gemm(const bf16_t* __restrict__ h1, const bf16_t* __restrict__ w2,
                         const float* __restrict__ fc2_b, const float* __restrict__ x1,
                         float* __restrict__ out) {
    int lane = threadIdx.x & 31;
    int wave = threadIdx.x >> 5;
    int tid  = blockIdx.x * 8 + wave;
    const int TN = kC / 32;                 // 6
    int tm = tid / TN, tn = tid % TN;

    Acc2x2 acc = gemm_2x2<24>(h1 + tm * 32 * 768, 768, w2 + tn * 32 * 768, 768, lane);

    int ncol = lane & 15, hi8 = (lane >> 4) * 8;
#pragma unroll
    for (int ni = 0; ni < 2; ++ni) {
        int col = tn * 32 + ni * 16 + ncol;
        float bias = fc2_b[col];
#pragma unroll
        for (int mi = 0; mi < 2; ++mi) {
#pragma unroll
            for (int r = 0; r < 8; ++r) {
                int row = tm * 32 + mi * 16 + r + hi8;
                out[row * kC + col] = x1[row * kC + col] + acc.m[mi][ni][r] + bias;
            }
        }
    }
}

// ---------------- host launcher ----------------
extern "C" void kernel_launch(void* const* d_in, const int* in_sizes, int n_in,
                              void* d_out, int out_size, void* d_ws, size_t ws_size,
                              hipStream_t stream) {
    (void)in_sizes; (void)n_in; (void)out_size; (void)ws_size;
    const float* x      = (const float*)d_in[0];
    const float* n1g    = (const float*)d_in[1];
    const float* n1b    = (const float*)d_in[2];
    const float* qkv_w  = (const float*)d_in[3];
    const float* qkv_b  = (const float*)d_in[4];
    const float* rpb    = (const float*)d_in[5];
    const float* proj_w = (const float*)d_in[6];
    const float* proj_b = (const float*)d_in[7];
    const float* n2g    = (const float*)d_in[8];
    const float* n2b    = (const float*)d_in[9];
    const float* fc1_w  = (const float*)d_in[10];
    const float* fc1_b  = (const float*)d_in[11];
    const float* fc2_w  = (const float*)d_in[12];
    const float* fc2_b  = (const float*)d_in[13];
    float* out = (float*)d_out;

    char* ws = (char*)d_ws;
    size_t off = 0;
    auto take = [&](size_t bytes) -> char* {
        char* p = ws + off;
        off = (off + bytes + 255) & ~(size_t)255;
        return p;
    };
    bf16_t* wq   = (bf16_t*)take((size_t)576 * 192 * 2);
    bf16_t* wp   = (bf16_t*)take((size_t)192 * 192 * 2);
    bf16_t* w1   = (bf16_t*)take((size_t)768 * 192 * 2);
    bf16_t* w2   = (bf16_t*)take((size_t)192 * 768 * 2);
    bf16_t* xs   = (bf16_t*)take((size_t)kTOK * kC * 2);        // LN1, strip layout
    bf16_t* qb   = (bf16_t*)take((size_t)kNS * kNH * kN * kHD * 2);
    bf16_t* kbuf = (bf16_t*)take((size_t)kNS * kNH * kN * kHD * 2);
    bf16_t* vtb  = (bf16_t*)take((size_t)kNS * kNH * kHD * kN * 2);
    bf16_t* ob   = (bf16_t*)take((size_t)kTOK * kC * 2);        // attn out, strip layout
    float*  x1   = (float*) take((size_t)kTOK * kC * 4);        // post-attn residual
    bf16_t* xn2  = (bf16_t*)take((size_t)kTOK * kC * 2);        // LN2
    bf16_t* h1   = (bf16_t*)take((size_t)kTOK * 768 * 2);       // GELU(FC1)

    cvt_weights<<<576, 256, 0, stream>>>(qkv_w, proj_w, fc1_w, fc2_w, wq, wp, w1, w2);
    ln_kernel<1><<<kTOK / 8, 256, 0, stream>>>(x, n1g, n1b, xs);
    qkv_gemm<<<(kTOK / 32) * (576 / 32) / 8, 256, 0, stream>>>(xs, wq, qkv_b, qb, kbuf, vtb);
    attn_kernel<<<kNS * kNH * 16, 128, 0, stream>>>(qb, kbuf, vtb, rpb, ob);
    proj_gemm<<<(kTOK / 32) * (kC / 32) / 8, 256, 0, stream>>>(ob, wp, proj_b, x, x1);
    ln_kernel<0><<<kTOK / 8, 256, 0, stream>>>(x1, n2g, n2b, xn2);
    fc1_gemm<<<(kTOK / 32) * (768 / 32) / 8, 256, 0, stream>>>(xn2, w1, fc1_b, h1);
    fc2_gemm<<<(kTOK / 32) * (kC / 32) / 8, 256, 0, stream>>>(h1, w2, fc2_b, x1, out);
}
